// GraphNet_79224966742551
// MI455X (gfx1250) — matrix operand order repaired and checked by reading, not compiled, not run on hardware
//
#include <hip/hip_runtime.h>
#include <hip/hip_bf16.h>

typedef __attribute__((ext_vector_type(16))) _Float16 v16h;
typedef __attribute__((ext_vector_type(8)))  float    v8f;

#define N_NODES 8192
#define N_EDGES 262144
#define N_E_TOT (N_EDGES + N_NODES)
#define N_GRAPHS 64
#define BN_PAD_TOT 208   // padded channels: 32 + 48 + 64 + 64

// ---------------- workspace layout ----------------
constexpr size_t AL(size_t x){ return (x + 255) & ~size_t(255); }
constexpr size_t OFF_FRAG  = 0;                                    // f16 conv A-frags: 65536 halves
constexpr size_t OFF_LINB  = AL(OFF_FRAG + 65536*2);               // f16 lin_l/lin_r B-frags: 16384 halves
constexpr size_t OFF_BNA   = AL(OFF_LINB + 16384*2);               // f32[208] BN scale (padded)
constexpr size_t OFF_BNO   = AL(OFF_BNA + BN_PAD_TOT*4);           // f32[208] BN offset (padded)
constexpr size_t OFF_X     = AL(OFF_BNO + BN_PAD_TOT*4);           // f32[N*64]   CNN features
constexpr size_t OFF_XL    = AL(OFF_X   + (size_t)N_NODES*64*4);   // f32[N*128]
constexpr size_t OFF_XR    = AL(OFF_XL  + (size_t)N_NODES*128*4);  // f32[N*128]
constexpr size_t OFF_CNT   = AL(OFF_XR  + (size_t)N_NODES*128*4);  // i32[N]
constexpr size_t OFF_ESUM  = AL(OFF_CNT + N_NODES*4);              // f32[N]
constexpr size_t OFF_LOOP  = AL(OFF_ESUM+ N_NODES*4);              // f32[N]
constexpr size_t OFF_OFFS  = AL(OFF_LOOP+ N_NODES*4);              // i32[N+1]
constexpr size_t OFF_FILL  = AL(OFF_OFFS+ (N_NODES+1)*4);          // i32[N]
constexpr size_t OFF_CSR   = AL(OFF_FILL+ N_NODES*4);              // i32[E+N]
constexpr size_t OFF_SCORE = AL(OFF_CSR + (size_t)N_E_TOT*4);      // f32[(E+N)*2]
constexpr size_t OFF_X2    = AL(OFF_SCORE + (size_t)N_E_TOT*2*4);  // f32[N*128]
constexpr size_t OFF_GATE  = AL(OFF_X2  + (size_t)N_NODES*128*4);  // f32[N]
constexpr size_t OFF_GM    = AL(OFF_GATE+ N_NODES*4);              // f32[64]
constexpr size_t OFF_GD    = AL(OFF_GM  + 64*4);                   // f32[64]
constexpr size_t OFF_POOL  = AL(OFF_GD  + 64*4);                   // f32[64*128]

// ================= weight / fragment packing =================
struct PackArgs {
  const float* w[4];
  const float* cb[4]; const float* g[4]; const float* bb[4];
  const float* m[4];  const float* v[4];
  const float* linl;  const float* linr;
};

__global__ void k_pack(PackArgs pa, _Float16* __restrict__ frag, _Float16* __restrict__ linb,
                       float* __restrict__ abn, float* __restrict__ obn) {
  const int tid = threadIdx.x; const int nt = blockDim.x;
  const int CIN[4]={3,18,36,54}, COUT[4]={18,36,54,54}, MT[4]={2,3,4,4}, KT[4]={1,6,11,16};
  const int FB[4]={0,1024,10240,32768};
  const int BNP[4]={0,32,80,144};   // padded BN bases (MT*16 channels per layer)
  for (int l=0;l<4;l++){
    const float* W = pa.w[l];
    const int total = MT[l]*KT[l]*512;
    for (int idx=tid; idx<total; idx+=nt){
      int j = idx & 15, lane = (idx>>4)&31, fr = idx>>9;
      int kt = fr % KT[l], mt = fr / KT[l];
      int co = mt*16 + (lane & 15);
      // ISA A-fragment (16-bit, 16x32): lane holds M=lane&15; K = (j<8?j:j+8) + 8*(lane>=16)
      int K  = kt*32 + ((lane>>4)?8:0) + (j<8 ? j : j+8);
      float val = 0.f;
      if (co < COUT[l] && K < CIN[l]*9) val = W[co*CIN[l]*9 + K];
      frag[FB[l] + idx] = (_Float16)val;
    }
    for (int c=tid;c<MT[l]*16;c+=nt){
      float a = 0.f, o = 0.f;
      if (c < COUT[l]) {
        a = pa.g[l][c] * rsqrtf(pa.v[l][c] + 1e-5f);
        o = (pa.cb[l][c] - pa.m[l][c]) * a + pa.bb[l][c];
      }
      abn[BNP[l]+c] = a;
      obn[BNP[l]+c] = o;
    }
  }
  for (int which=0; which<2; ++which){
    const float* W = which ? pa.linr : pa.linl;   // [128,64]
    for (int idx=tid; idx<8192; idx+=nt){
      int j = idx & 15, lane = (idx>>4)&31, fr = idx>>9;  // fr = kt*8 + ntile
      int ntl = fr & 7, kt = fr >> 3;
      // B-fragment (16-bit, 32x16): lane holds N=lane&15; K = j + 16*(lane>=16)
      int K = kt*32 + ((lane>>4)<<4) + j;
      int n = ntl*16 + (lane & 15);
      linb[which*8192 + idx] = (_Float16)W[n*64 + K];
    }
  }
}

// ================= init =================
__global__ void k_init(int* cnt, float* esum, int* fill, float* gd, float* gm, float* pooled){
  int i = blockIdx.x*blockDim.x + threadIdx.x;
  if (i < N_NODES){ cnt[i]=0; esum[i]=0.f; fill[i]=0; pooled[i]=0.f; } // pooled: 64*128 == 8192
  if (i < N_GRAPHS){ gd[i]=0.f; gm[i]=-3.0e38f; }
}

// ================= conv stack (implicit GEMM via WMMA) =================
// Wave w owns pixel-tile pt=w and keeps MT accumulators, so the (expensive)
// B-fragment gather is built ONCE per (ptile, ktile) and reused across all
// MT A-tiles. Branch-free gathers (clamped address + select).
template<int CIN,int HIN,int WIN,int COUT,int CR,int MT,int KT,int FRAGBASE,int BNPAD>
__device__ __forceinline__ void conv_layer(const _Float16* IN, _Float16* OUT, _Float16* unp,
    const _Float16* __restrict__ frags, const float* s_abn, const float* s_obn)
{
  const int tid  = threadIdx.x;
  const int lane = tid & 31;
  const int wv   = tid >> 5;
  constexpr int PT   = (CR*WIN + 15) / 16;   // pixel tiles per chunk (<= 4)
  constexpr int PTW  = PT * 16;              // padded chunk width (unp row stride)
  constexpr int NCH  = HIN / CR;
  constexpr int K9   = CIN * 9;
  const int n  = lane & 15;
  const int kb = (lane >> 4) << 4;
  const int lanehi = lane >> 4;

  for (int ch = 0; ch < NCH; ++ch) {
    const int r0 = ch * CR;
    if (wv < PT) {                     // wave-uniform; no barriers inside
      const int pt = wv;
      const int plocal = pt*16 + n;
      const bool pvalid = plocal < CR*WIN;
      const int p  = r0*WIN + plocal;
      const int oy = p / WIN;          // WIN is power of two
      const int ox = p & (WIN-1);
      v8f acc[MT];
      #pragma unroll
      for (int mt = 0; mt < MT; ++mt) acc[mt] = (v8f){};
      for (int kt = 0; kt < KT; ++kt) {
        v16h bfrag;
        #pragma unroll
        for (int j = 0; j < 16; ++j) {
          int K = kt*32 + (lanehi ? 8 : 0) + (j < 8 ? j : j+8);
          int ci  = K / 9;
          int rem = K - ci*9;
          int dy  = rem / 3;
          int dx  = rem - dy*3;
          int iy = oy + dy - 1, ix = ox + dx - 1;
          bool ok = pvalid & (K < K9) & (iy >= 0) & (iy < HIN) & ((unsigned)ix < (unsigned)WIN);
          int addr = ok ? (ci*HIN*WIN + iy*WIN + ix) : 0;   // 0 is always a safe LDS index
          float val = (float)IN[addr];
          bfrag[j] = (_Float16)(ok ? val : 0.f);
        }
        #pragma unroll
        for (int mt = 0; mt < MT; ++mt) {
          v16h afrag = *(const v16h*)(frags + FRAGBASE + ((size_t)(mt*KT + kt)*32 + lane)*16);
          acc[mt] = __builtin_amdgcn_wmma_f32_16x16x32_f16(false, afrag, false, bfrag,
                                                           (short)0, acc[mt], false, false);
        }
      }
      // BN + ReLU + store (unconditional: unp padded to MT*16 x PTW)
      #pragma unroll
      for (int mt = 0; mt < MT; ++mt) {
        #pragma unroll
        for (int q = 0; q < 8; ++q) {
          int m  = q + (lanehi << 3);
          int co = mt*16 + m;
          float f = fmaxf(acc[mt][q] * s_abn[BNPAD+co] + s_obn[BNPAD+co], 0.f);
          unp[co*PTW + plocal] = (_Float16)f;
        }
      }
    }
    __syncthreads();
    // 2x2 max pool of this chunk (reads only valid region)
    constexpr int WP = WIN/2;
    constexpr int PE = COUT * (CR/2) * WP;
    for (int t = tid; t < PE; t += 128) {
      int co = t / ((CR/2)*WP);
      int r  = t - co*((CR/2)*WP);
      int yy = r / WP, xx = r - yy*WP;
      const _Float16* u = unp + co*PTW + (2*yy)*WIN + 2*xx;
      float a0=(float)u[0], a1=(float)u[1], a2=(float)u[WIN], a3=(float)u[WIN+1];
      OUT[co*(HIN/2)*WP + (r0/2 + yy)*WP + xx] = (_Float16)fmaxf(fmaxf(a0,a1), fmaxf(a2,a3));
    }
    __syncthreads();
  }
}

__global__ __launch_bounds__(128) void k_conv(const float* __restrict__ signal,
    const _Float16* __restrict__ frags, const float* __restrict__ abn, const float* __restrict__ obn,
    const float* __restrict__ fcw, const float* __restrict__ fcb, float* __restrict__ x)
{
  __shared__ _Float16 bufA[4608];   // signal(4608) -> L2out(3456) -> L4out(324)
  __shared__ _Float16 bufB[6912];   // L1out(6912) -> L3out(1296)
  __shared__ _Float16 unp[3072];    // per-chunk unpooled scratch (padded MT*16 x PT*16)
  __shared__ float s_abn[BN_PAD_TOT], s_obn[BN_PAD_TOT];
  const int node = blockIdx.x;
  const int tid  = threadIdx.x;

  for (int t = tid; t < BN_PAD_TOT; t += 128){ s_abn[t] = abn[t]; s_obn[t] = obn[t]; }
  const float* in = signal + (size_t)node * 4608;
  for (int t = tid; t < 4608; t += 128) bufA[t] = (_Float16)in[t];
  __syncthreads();

  conv_layer< 3,48,32,18,2,2, 1,    0,  0>(bufA, bufB, unp, frags, s_abn, s_obn);  // PT=4
  conv_layer<18,24,16,36,4,3, 6, 1024, 32>(bufB, bufA, unp, frags, s_abn, s_obn);  // PT=4
  conv_layer<36,12, 8,54,6,4,11,10240, 80>(bufA, bufB, unp, frags, s_abn, s_obn);  // PT=3
  conv_layer<54, 6, 4,54,6,4,16,32768,144>(bufB, bufA, unp, frags, s_abn, s_obn);  // PT=2

  // FC 324 -> 64
  if (tid < 64) {
    const float* fw = fcw + tid*324;
    float s = fcb[tid];
    for (int i = 0; i < 324; ++i) s += fw[i] * (float)bufA[i];
    x[(size_t)node*64 + tid] = s;
  }
}

// ================= x_l / x_r GEMM via WMMA =================
__global__ __launch_bounds__(128) void k_xlr(const float* __restrict__ x,
    const _Float16* __restrict__ linb, const float* __restrict__ bl, const float* __restrict__ br,
    float* __restrict__ xl, float* __restrict__ xr)
{
  const int node0 = blockIdx.x * 16;
  const int tid = threadIdx.x;
  const int lane = tid & 31, wv = tid >> 5;
  const int lanehi = lane >> 4;
  for (int i = 0; i < 4; ++i) {
    const int combo = wv*4 + i;         // 0..15  (wave-uniform)
    const int which = combo >> 3, ntl = combo & 7;
    v8f acc = {};
    for (int kt = 0; kt < 2; ++kt) {
      v16h a;
      #pragma unroll
      for (int j = 0; j < 16; ++j) {
        int K = kt*32 + (lanehi << 4) + j;
        a[j] = (_Float16)x[(size_t)(node0 + (lane & 15))*64 + K];
      }
      v16h b = *(const v16h*)(linb + which*8192 + ((kt*8 + ntl)*32 + lane)*16);
      acc = __builtin_amdgcn_wmma_f32_16x16x32_f16(false, a, false, b, (short)0, acc, false, false);
    }
    const float* bias = which ? br : bl;
    float* out = which ? xr : xl;
    #pragma unroll
    for (int q = 0; q < 8; ++q) {
      int m = q + (lanehi << 3);
      int ncol = ntl*16 + (lane & 15);
      out[(size_t)(node0 + m)*128 + ncol] = acc[q] + bias[ncol];
    }
  }
}

// ================= CSR build =================
__global__ void k_cnt(const int* __restrict__ ei, const float* __restrict__ ew,
                      int* __restrict__ cnt, float* __restrict__ esum){
  int e = blockIdx.x*blockDim.x + threadIdx.x;
  if (e < N_EDGES){
    int t = ei[N_EDGES + e];
    atomicAdd(&cnt[t], 1);
    atomicAdd(&esum[t], ew[e]);
  }
}

__global__ __launch_bounds__(1024) void k_scan(const int* __restrict__ cnt, const float* __restrict__ esum,
                       float* __restrict__ loop_attr, int* __restrict__ offs){
  __shared__ int ps[1024];
  const int tid = threadIdx.x;
  const int base = tid*8;
  int loc[8]; int s = 0;
  for (int k=0;k<8;k++){ loc[k] = cnt[base+k] + 1; s += loc[k]; }
  ps[tid] = s; __syncthreads();
  for (int off=1; off<1024; off<<=1){
    int v = 0;
    if (tid >= off) v = ps[tid-off];
    __syncthreads();
    ps[tid] += v;
    __syncthreads();
  }
  int run = ps[tid] - s;  // exclusive prefix
  for (int k=0;k<8;k++){ offs[base+k] = run; run += loc[k]; }
  if (tid == 1023) offs[N_NODES] = run;   // == E+N
  for (int n=tid; n<N_NODES; n+=1024)
    loop_attr[n] = esum[n] / fmaxf((float)cnt[n], 1.0f);
}

__global__ void k_fill(const int* __restrict__ ei, const int* __restrict__ offs,
                       int* __restrict__ fill, int* __restrict__ csr){
  int i = blockIdx.x*blockDim.x + threadIdx.x;
  if (i < N_EDGES){
    int t = ei[N_EDGES + i];
    int pos = atomicAdd(&fill[t], 1);
    csr[offs[t] + pos] = i;
  } else if (i < N_E_TOT){
    int n = i - N_EDGES;
    csr[offs[n+1] - 1] = N_EDGES + n;     // self loop in last slot
  }
}

// ================= block reductions =================
__device__ __forceinline__ float blk_sum(float v, float* red){
  int tid = threadIdx.x;
  red[tid] = v; __syncthreads();
  #pragma unroll
  for (int s=64; s>0; s>>=1){ if (tid < s) red[tid] += red[tid+s]; __syncthreads(); }
  float r = red[0]; __syncthreads(); return r;
}
__device__ __forceinline__ float blk_max(float v, float* red){
  int tid = threadIdx.x;
  red[tid] = v; __syncthreads();
  #pragma unroll
  for (int s=64; s>0; s>>=1){ if (tid < s) red[tid] = fmaxf(red[tid], red[tid+s]); __syncthreads(); }
  float r = red[0]; __syncthreads(); return r;
}

// ================= GATv2 per-node softmax + aggregate + LN + gate =================
__global__ __launch_bounds__(128) void k_gat(const float* __restrict__ xl, const float* __restrict__ xr,
    const int* __restrict__ ei, const float* __restrict__ ew, const float* __restrict__ loop_attr,
    const int* __restrict__ offs, const int* __restrict__ csr, float* __restrict__ scores,
    const float* __restrict__ we, const float* __restrict__ attv, const float* __restrict__ gbias,
    const float* __restrict__ lng, const float* __restrict__ lnb,
    const float* __restrict__ gatew, const float* __restrict__ gateb,
    float* __restrict__ x2, float* __restrict__ gate)
{
  const int n = blockIdx.x, tid = threadIdx.x;
  __shared__ float red[128];
  __shared__ float sh_xr[128], sh_we[128], sh_att[128];
  const int nb = offs[n], ne = offs[n+1];
  sh_xr[tid] = xr[(size_t)n*128 + tid];
  sh_we[tid] = we[tid];
  sh_att[tid] = attv[tid];
  __syncthreads();

  // ---- scores + running max ----
  float lm0 = -3.0e38f, lm1 = -3.0e38f;
  for (int pos = nb + tid; pos < ne; pos += 128) {
    int eid = csr[pos]; int src; float attr;
    if (eid < N_EDGES){ src = ei[eid]; attr = ew[eid]; }
    else              { src = eid - N_EDGES; attr = loop_attr[src]; }
    const float* xls = xl + (size_t)src*128;
    float s0 = 0.f, s1 = 0.f;
    for (int c = 0; c < 64; ++c){
      float z = xls[c] + sh_xr[c] + attr*sh_we[c];
      z = z > 0.f ? z : 0.2f*z;
      s0 += z * sh_att[c];
    }
    for (int c = 64; c < 128; ++c){
      float z = xls[c] + sh_xr[c] + attr*sh_we[c];
      z = z > 0.f ? z : 0.2f*z;
      s1 += z * sh_att[c];
    }
    scores[(size_t)pos*2+0] = s0; scores[(size_t)pos*2+1] = s1;
    lm0 = fmaxf(lm0, s0); lm1 = fmaxf(lm1, s1);
  }
  float m0 = blk_max(lm0, red);
  float m1 = blk_max(lm1, red);

  // ---- exp sums ----
  float ls0 = 0.f, ls1 = 0.f;
  for (int pos = nb + tid; pos < ne; pos += 128) {
    ls0 += __expf(scores[(size_t)pos*2+0] - m0);
    ls1 += __expf(scores[(size_t)pos*2+1] - m1);
  }
  float d0 = blk_sum(ls0, red);
  float d1 = blk_sum(ls1, red);

  // ---- normalize to alpha in-place ----
  for (int pos = nb + tid; pos < ne; pos += 128) {
    scores[(size_t)pos*2+0] = __expf(scores[(size_t)pos*2+0] - m0) / d0;
    scores[(size_t)pos*2+1] = __expf(scores[(size_t)pos*2+1] - m1) / d1;
  }
  __syncthreads();

  // ---- aggregate: thread = channel ----
  const int h = tid >> 6;
  float acc = 0.f;
  for (int pos = nb; pos < ne; ++pos) {
    int eid = csr[pos];
    int src = eid < N_EDGES ? ei[eid] : eid - N_EDGES;
    acc += scores[(size_t)pos*2 + h] * xl[(size_t)src*128 + tid];
  }
  float o = acc + gbias[tid];

  // ---- LayerNorm + ReLU ----
  float mu  = blk_sum(o, red) * (1.0f/128.0f);
  float dev = o - mu;
  float var = blk_sum(dev*dev, red) * (1.0f/128.0f);
  float y = dev * rsqrtf(var + 1e-5f) * lng[tid] + lnb[tid];
  y = fmaxf(y, 0.f);
  x2[(size_t)n*128 + tid] = y;

  // ---- gate scalar ----
  float gs = blk_sum(y * gatew[tid], red);
  if (tid == 0) gate[n] = fmaxf(gs + gateb[0], 0.f);
}

// ================= graph pooling =================
__global__ void k_pool1(const float* __restrict__ gate, const int* __restrict__ batch, float* gm){
  int n = blockIdx.x*blockDim.x + threadIdx.x;
  if (n >= N_NODES) return;
  float v = gate[n]; int b = batch[n];
  if (v >= 0.f) atomicMax((int*)(gm + b), __float_as_int(v));
  else          atomicMin((unsigned int*)(gm + b), __float_as_uint(v));
}
__global__ void k_pool2(const float* __restrict__ gate, const int* __restrict__ batch,
                        const float* __restrict__ gm, float* gd){
  int n = blockIdx.x*blockDim.x + threadIdx.x;
  if (n >= N_NODES) return;
  int b = batch[n];
  atomicAdd(&gd[b], __expf(gate[n] - gm[b]));
}
__global__ void k_pool3(const float* __restrict__ x2, const float* __restrict__ gate,
                        const int* __restrict__ batch, const float* __restrict__ gm,
                        const float* __restrict__ gd, float* pooled){
  int i = blockIdx.x*blockDim.x + threadIdx.x;
  if (i >= N_NODES*128) return;
  int n = i >> 7, c = i & 127;
  int b = batch[n];
  float a = __expf(gate[n] - gm[b]) / gd[b];
  atomicAdd(&pooled[b*128 + c], x2[i] * a);
}
__global__ void k_out(const float* __restrict__ pooled, const float* __restrict__ outw,
                      const float* __restrict__ outb, float* __restrict__ dout){
  int t = threadIdx.x;
  if (t >= 128) return;
  int g = t >> 1, k = t & 1;
  float s = outb[k];
  for (int c = 0; c < 128; ++c) s += pooled[g*128 + c] * outw[k*128 + c];
  dout[g*2 + k] = s;
}

// ================= launch =================
extern "C" void kernel_launch(void* const* d_in, const int* in_sizes, int n_in,
                              void* d_out, int out_size, void* d_ws, size_t ws_size,
                              hipStream_t stream) {
  (void)in_sizes; (void)n_in; (void)out_size; (void)ws_size;
  const float* signal = (const float*)d_in[2];
  const int*   eidx   = (const int*)d_in[4];     // [2][E]
  const float* ew     = (const float*)d_in[5];   // [E]
  const int*   batch  = (const int*)d_in[6];
  // params (setup_inputs dict insertion order, flattened)
  const float* fcw   = (const float*)d_in[31];
  const float* fcb   = (const float*)d_in[32];
  const float* linlb = (const float*)d_in[34];
  const float* linrb = (const float*)d_in[36];
  const float* we    = (const float*)d_in[37];
  const float* attv  = (const float*)d_in[38];
  const float* gbias = (const float*)d_in[39];
  const float* lng   = (const float*)d_in[40];
  const float* lnb   = (const float*)d_in[41];
  const float* gatew = (const float*)d_in[42];
  const float* gateb = (const float*)d_in[43];
  const float* outw  = (const float*)d_in[44];
  const float* outb  = (const float*)d_in[45];

  char* ws = (char*)d_ws;
  _Float16* frag = (_Float16*)(ws + OFF_FRAG);
  _Float16* linb = (_Float16*)(ws + OFF_LINB);
  float* abn  = (float*)(ws + OFF_BNA);
  float* obn  = (float*)(ws + OFF_BNO);
  float* x    = (float*)(ws + OFF_X);
  float* xl   = (float*)(ws + OFF_XL);
  float* xr   = (float*)(ws + OFF_XR);
  int*   cnt  = (int*)  (ws + OFF_CNT);
  float* esum = (float*)(ws + OFF_ESUM);
  float* loop = (float*)(ws + OFF_LOOP);
  int*   offs = (int*)  (ws + OFF_OFFS);
  int*   fill = (int*)  (ws + OFF_FILL);
  int*   csr  = (int*)  (ws + OFF_CSR);
  float* scr  = (float*)(ws + OFF_SCORE);
  float* x2   = (float*)(ws + OFF_X2);
  float* gate = (float*)(ws + OFF_GATE);
  float* gm   = (float*)(ws + OFF_GM);
  float* gd   = (float*)(ws + OFF_GD);
  float* pool = (float*)(ws + OFF_POOL);

  PackArgs pa;
  for (int l = 0; l < 4; ++l) {
    pa.w[l]  = (const float*)d_in[7 + 6*l];
    pa.cb[l] = (const float*)d_in[8 + 6*l];
    pa.g[l]  = (const float*)d_in[9 + 6*l];
    pa.bb[l] = (const float*)d_in[10 + 6*l];
    pa.m[l]  = (const float*)d_in[11 + 6*l];
    pa.v[l]  = (const float*)d_in[12 + 6*l];
  }
  pa.linl = (const float*)d_in[33];
  pa.linr = (const float*)d_in[35];

  k_pack<<<1, 256, 0, stream>>>(pa, frag, linb, abn, obn);
  k_init<<<(N_NODES + 255)/256, 256, 0, stream>>>(cnt, esum, fill, gd, gm, pool);
  k_conv<<<N_NODES, 128, 0, stream>>>(signal, frag, abn, obn, fcw, fcb, x);
  k_cnt<<<N_EDGES/256, 256, 0, stream>>>(eidx, ew, cnt, esum);
  k_scan<<<1, 1024, 0, stream>>>(cnt, esum, loop, offs);
  k_fill<<<(N_E_TOT + 255)/256, 256, 0, stream>>>(eidx, offs, fill, csr);
  k_xlr<<<N_NODES/16, 128, 0, stream>>>(x, linb, linlb, linrb, xl, xr);
  k_gat<<<N_NODES, 128, 0, stream>>>(xl, xr, eidx, ew, loop, offs, csr, scr,
                                     we, attv, gbias, lng, lnb, gatew, gateb, x2, gate);
  k_pool1<<<(N_NODES + 255)/256, 256, 0, stream>>>(gate, batch, gm);
  k_pool2<<<(N_NODES + 255)/256, 256, 0, stream>>>(gate, batch, gm, gd);
  k_pool3<<<(N_NODES*128)/256, 256, 0, stream>>>(x2, gate, batch, gm, gd, pool);
  k_out<<<1, 128, 0, stream>>>(pool, outw, outb, (float*)d_out);
}
